// EnhancedMultiHeadAttention_62749472195144
// MI455X (gfx1250) — matrix operand order repaired and checked by reading, not compiled, not run on hardware
//
#include <hip/hip_runtime.h>
#include <hip/hip_bf16.h>
#include <math.h>

// ---------------------------------------------------------------------------
// Problem constants (match the reference)
// ---------------------------------------------------------------------------
#define B_ 2
#define S_ 2048
#define D_ 1024
#define H_ 16
#define Dh_ 64
#define M_ (B_ * S_)          // 4096 tokens
#define LN_EPS 1e-5f
#define SC_LDR 2052           // LDS score row stride (floats); 2052 % 64 == 4 -> no bank conflicts

typedef __bf16 bf16_t;
typedef __attribute__((ext_vector_type(16))) __bf16 v16bf;
typedef __attribute__((ext_vector_type(8)))  float  v8f;

// ---------------------------------------------------------------------------
// WMMA fragment loaders (CDNA5 wave32 16x16x32 bf16 layouts, per ISA 7.12.2)
// ---------------------------------------------------------------------------
// A matrix 16x32 (MxK), row-major source A[row*lda + k]:
//  lane<16 : row = lane,    elems 0-7 = K k0..k0+7,    elems 8-15 = K k0+16..k0+23
//  lane>=16: row = lane-16, elems 0-7 = K k0+8..k0+15, elems 8-15 = K k0+24..k0+31
__device__ __forceinline__ v16bf load_a_frag_bf16(const bf16_t* __restrict__ A,
                                                  int lda, int m_base, int k0, int lane) {
    int row = m_base + (lane & 15);
    const bf16_t* p = A + (size_t)row * lda + k0 + ((lane & 16) ? 8 : 0);
    union { v16bf v; int4 q[2]; } u;
    u.q[0] = *(const int4*)(p);
    u.q[1] = *(const int4*)(p + 16);
    return u.v;
}

// B matrix 32x16 (KxN) sourced from a [N][K]-layout buffer Bt (B[k][n] = Bt[n*ldb+k]):
//  lane<16 : col = lane,    elems 0-7 = K k0..k0+7,    elems 8-15 = K k0+16..k0+23
//  lane>=16: col = lane-16, elems 0-7 = K k0+8..k0+15, elems 8-15 = K k0+24..k0+31
__device__ __forceinline__ v16bf load_b_frag_bf16(const bf16_t* __restrict__ Bt,
                                                  int ldb, int n_base, int k0, int lane) {
    int col = n_base + (lane & 15);
    const bf16_t* p = Bt + (size_t)col * ldb + k0 + ((lane & 16) ? 8 : 0);
    union { v16bf v; int4 q[2]; } u;
    u.q[0] = *(const int4*)(p);
    u.q[1] = *(const int4*)(p + 16);
    return u.v;
}

// A fragment from fp32 source (LDS prob tile), converted to bf16 in registers.
__device__ __forceinline__ v16bf cvt_a_frag_f32(const float* p) {
    float4 x0 = *(const float4*)(p);
    float4 x1 = *(const float4*)(p + 4);
    float4 x2 = *(const float4*)(p + 16);
    float4 x3 = *(const float4*)(p + 20);
    v16bf v;
    v[0]=(bf16_t)x0.x; v[1]=(bf16_t)x0.y; v[2]=(bf16_t)x0.z; v[3]=(bf16_t)x0.w;
    v[4]=(bf16_t)x1.x; v[5]=(bf16_t)x1.y; v[6]=(bf16_t)x1.z; v[7]=(bf16_t)x1.w;
    v[8]=(bf16_t)x2.x; v[9]=(bf16_t)x2.y; v[10]=(bf16_t)x2.z; v[11]=(bf16_t)x2.w;
    v[12]=(bf16_t)x3.x; v[13]=(bf16_t)x3.y; v[14]=(bf16_t)x3.z; v[15]=(bf16_t)x3.w;
    return v;
}

// ---------------------------------------------------------------------------
// Conversion / transpose kernels
// ---------------------------------------------------------------------------
__global__ void cvt_f32_to_bf16_kernel(const float* __restrict__ src,
                                       bf16_t* __restrict__ dst, int n) {
    int i = blockIdx.x * blockDim.x + threadIdx.x;
    int stride = gridDim.x * blockDim.x;
    for (; i < n; i += stride) dst[i] = (bf16_t)src[i];
}

// Wt[n*D + k] = (bf16) W[k*D + n]   (D_ x D_ matrix)
__global__ __launch_bounds__(256) void transpose_cvt_kernel(const float* __restrict__ W,
                                                            bf16_t* __restrict__ Wt) {
    __shared__ float tile[32][33];
    int x = blockIdx.x * 32 + threadIdx.x;
    int y = blockIdx.y * 32 + threadIdx.y;
    #pragma unroll
    for (int i = 0; i < 32; i += 8)
        tile[threadIdx.y + i][threadIdx.x] = W[(size_t)(y + i) * D_ + x];
    __syncthreads();
    int xo = blockIdx.y * 32 + threadIdx.x;
    int yo = blockIdx.x * 32 + threadIdx.y;
    #pragma unroll
    for (int i = 0; i < 32; i += 8)
        Wt[(size_t)(yo + i) * D_ + xo] = (bf16_t)tile[threadIdx.x][threadIdx.y + i];
}

// ---------------------------------------------------------------------------
// Projection GEMM: Y[m][n] = sum_k A[m][k] * W[k][n] + bias[n]
//   mode 0: bf16 -> dst[B][H][S][Dh]  (Q, K)
//   mode 1: bf16 -> dst[B][H][Dh][S]  (V transposed)
//   mode 2: fp32 -> dst[m*N + n]      (output projection, pre-LN)
// ---------------------------------------------------------------------------
__global__ __launch_bounds__(256) void proj_gemm_kernel(
    const bf16_t* __restrict__ A, const bf16_t* __restrict__ Bt,
    const float* __restrict__ bias, void* __restrict__ out,
    int Mdim, int Ndim, int Kdim, int mode) {
    const int lane = threadIdx.x & 31;
    const int wid  = threadIdx.x >> 5;
    const int m_wave = blockIdx.y * 128 + (wid & 3) * 32;
    const int n_wave = blockIdx.x * 128 + (wid >> 2) * 64;

    v8f acc[2][4];
    #pragma unroll
    for (int i = 0; i < 2; ++i)
        #pragma unroll
        for (int j = 0; j < 4; ++j) acc[i][j] = 0.0f;

    for (int k0 = 0; k0 < Kdim; k0 += 32) {
        v16bf a0 = load_a_frag_bf16(A, Kdim, m_wave,      k0, lane);
        v16bf a1 = load_a_frag_bf16(A, Kdim, m_wave + 16, k0, lane);
        v16bf bfr[4];
        #pragma unroll
        for (int nt = 0; nt < 4; ++nt)
            bfr[nt] = load_b_frag_bf16(Bt, Kdim, n_wave + nt * 16, k0, lane);
        #pragma unroll
        for (int nt = 0; nt < 4; ++nt) {
            acc[0][nt] = __builtin_amdgcn_wmma_f32_16x16x32_bf16(
                false, a0, false, bfr[nt], (short)0, acc[0][nt], false, false);
            acc[1][nt] = __builtin_amdgcn_wmma_f32_16x16x32_bf16(
                false, a1, false, bfr[nt], (short)0, acc[1][nt], false, false);
        }
    }

    #pragma unroll
    for (int mt = 0; mt < 2; ++mt) {
        #pragma unroll
        for (int nt = 0; nt < 4; ++nt) {
            int n = n_wave + nt * 16 + (lane & 15);
            int m0 = m_wave + mt * 16 + ((lane & 16) ? 8 : 0);
            float bn = bias[n];
            #pragma unroll
            for (int i = 0; i < 8; ++i) {
                int m = m0 + i;
                float val = acc[mt][nt][i] + bn;
                if (mode == 2) {
                    ((float*)out)[(size_t)m * Ndim + n] = val;
                } else {
                    int bb = m >> 11;
                    int s  = m & (S_ - 1);
                    int h  = n >> 6;
                    int dh = n & (Dh_ - 1);
                    if (mode == 0)
                        ((bf16_t*)out)[(((size_t)bb * H_ + h) * S_ + s) * Dh_ + dh] = (bf16_t)val;
                    else
                        ((bf16_t*)out)[(((size_t)bb * H_ + h) * Dh_ + dh) * S_ + s] = (bf16_t)val;
                }
            }
        }
    }
}

// ---------------------------------------------------------------------------
// Fused attention: scores -> softmax -> probs out (NT stores) -> ctx = P.V
// One block = one (b,h) and 16 query rows. 256 threads = 8 wave32.
//   Phase A: wave w computes 16x256 score slice (K=64) into LDS (fp32).
//   Phase B: exact row softmax (shfl reductions); probs -> LDS + d_out (NT).
//   Phase C: ctx with WMMA; A-frags from LDS (fp32->bf16), K split across
//            wave pairs, partials combined through LDS.
// LDS: 16*2052*4 = 131,328 B scores + 8 KB partials (< 320 KB WGP limit).
// ---------------------------------------------------------------------------
__global__ __launch_bounds__(256) void fused_attn_kernel(
    const bf16_t* __restrict__ Qp, const bf16_t* __restrict__ Kp,
    const bf16_t* __restrict__ Vt, const int* __restrict__ mask,
    float* __restrict__ attn_out, bf16_t* __restrict__ ctx) {
    __shared__ float sc[16][SC_LDR];
    __shared__ float part[8 * 32 * 8];

    const int lane = threadIdx.x & 31;
    const int wid  = threadIdx.x >> 5;
    const int q0 = blockIdx.x * 16;
    const int bh = blockIdx.y;
    const int bb = bh >> 4, h = bh & 15;
    const bf16_t* Qh = Qp + (size_t)bh * S_ * Dh_;
    const bf16_t* Kh = Kp + (size_t)bh * S_ * Dh_;
    const bf16_t* Vh = Vt + (size_t)bh * Dh_ * S_;

    // ---- Phase A: scores ----
    v16bf a0 = load_a_frag_bf16(Qh, Dh_, q0, 0,  lane);
    v16bf a1 = load_a_frag_bf16(Qh, Dh_, q0, 32, lane);
    #pragma unroll
    for (int c = 0; c < 4; ++c) {
        const int ncol0 = wid * 256 + c * 64;
        v8f acc[4];
        #pragma unroll
        for (int nt = 0; nt < 4; ++nt) acc[nt] = 0.0f;
        #pragma unroll
        for (int nt = 0; nt < 4; ++nt) {
            v16bf b0 = load_b_frag_bf16(Kh, Dh_, ncol0 + nt * 16, 0,  lane);
            v16bf b1 = load_b_frag_bf16(Kh, Dh_, ncol0 + nt * 16, 32, lane);
            acc[nt] = __builtin_amdgcn_wmma_f32_16x16x32_bf16(
                false, a0, false, b0, (short)0, acc[nt], false, false);
            acc[nt] = __builtin_amdgcn_wmma_f32_16x16x32_bf16(
                false, a1, false, b1, (short)0, acc[nt], false, false);
        }
        #pragma unroll
        for (int nt = 0; nt < 4; ++nt) {
            int n = ncol0 + nt * 16 + (lane & 15);
            bool masked = (mask[bb * S_ + n] == 0);
            int m0 = (lane & 16) ? 8 : 0;
            #pragma unroll
            for (int i = 0; i < 8; ++i) {
                float val = masked ? -INFINITY : acc[nt][i] * 0.125f;  // 1/sqrt(64)
                sc[m0 + i][n] = val;
            }
        }
    }
    __syncthreads();

    // ---- Phase B: exact softmax per row (2 rows per wave) ----
    #pragma unroll
    for (int rr = 0; rr < 2; ++rr) {
        const int r = wid * 2 + rr;
        float mx = -INFINITY;
        for (int j = lane; j < S_; j += 32) mx = fmaxf(mx, sc[r][j]);
        #pragma unroll
        for (int off = 16; off > 0; off >>= 1) mx = fmaxf(mx, __shfl_xor(mx, off, 32));
        float sum = 0.0f;
        for (int j = lane; j < S_; j += 32) {
            float e = __expf(sc[r][j] - mx);
            sc[r][j] = e;
            sum += e;
        }
        #pragma unroll
        for (int off = 16; off > 0; off >>= 1) sum += __shfl_xor(sum, off, 32);
        float inv = 1.0f / sum;
        float* gp = attn_out + ((size_t)bh * S_ + q0 + r) * S_;
        for (int j = lane; j < S_; j += 32) {
            float pv = sc[r][j] * inv;
            sc[r][j] = pv;
            __builtin_nontemporal_store(pv, gp + j);   // streamed, never re-read
        }
    }
    __syncthreads();

    // ---- Phase C: ctx = P . V  (16 x 64, K = 2048) ----
    // wave w: n-tile = w & 3, K half = w >> 2 ; partials combined via LDS.
    {
        const int nt = wid & 3;
        const int kh = wid >> 2;
        v8f acc = 0.0f;
        for (int k0 = kh * (S_ / 2); k0 < (kh + 1) * (S_ / 2); k0 += 32) {
            const float* ap = &sc[lane & 15][k0 + ((lane & 16) ? 8 : 0)];
            v16bf a = cvt_a_frag_f32(ap);
            v16bf b = load_b_frag_bf16(Vh, S_, nt * 16, k0, lane);
            acc = __builtin_amdgcn_wmma_f32_16x16x32_bf16(
                false, a, false, b, (short)0, acc, false, false);
        }
        float* pp = &part[(wid * 32 + lane) * 8];
        #pragma unroll
        for (int i = 0; i < 8; ++i) pp[i] = acc[i];
    }
    __syncthreads();
    if (wid < 4) {
        const int nt = wid;
        const float* p0 = &part[((wid)     * 32 + lane) * 8];
        const float* p1 = &part[((wid + 4) * 32 + lane) * 8];
        int n  = nt * 16 + (lane & 15);
        int m0 = (lane & 16) ? 8 : 0;
        #pragma unroll
        for (int i = 0; i < 8; ++i) {
            float val = p0[i] + p1[i];
            ctx[((size_t)bb * S_ + q0 + m0 + i) * D_ + h * Dh_ + n] = (bf16_t)val;
        }
    }
}

// ---------------------------------------------------------------------------
// LayerNorm(residual + preout) with gamma/beta; one block per token row.
// ---------------------------------------------------------------------------
__global__ __launch_bounds__(256) void ln_kernel(
    const float* __restrict__ preout, const float* __restrict__ resid,
    const float* __restrict__ gamma, const float* __restrict__ beta,
    float* __restrict__ out) {
    __shared__ float red[256];
    const size_t row = blockIdx.x;
    const float* a = preout + row * D_;
    const float* r = resid  + row * D_;
    const int t = threadIdx.x;
    float x[4]; float s = 0.0f;
    #pragma unroll
    for (int i = 0; i < 4; ++i) { x[i] = a[t + 256 * i] + r[t + 256 * i]; s += x[i]; }
    red[t] = s; __syncthreads();
    for (int st = 128; st > 0; st >>= 1) { if (t < st) red[t] += red[t + st]; __syncthreads(); }
    const float mu = red[0] * (1.0f / D_); __syncthreads();
    float v = 0.0f;
    #pragma unroll
    for (int i = 0; i < 4; ++i) { float d = x[i] - mu; v += d * d; }
    red[t] = v; __syncthreads();
    for (int st = 128; st > 0; st >>= 1) { if (t < st) red[t] += red[t + st]; __syncthreads(); }
    const float inv = rsqrtf(red[0] * (1.0f / D_) + LN_EPS);
    #pragma unroll
    for (int i = 0; i < 4; ++i) {
        int c = t + 256 * i;
        out[row * D_ + c] = (x[i] - mu) * inv * gamma[c] + beta[c];
    }
}

// ---------------------------------------------------------------------------
// Host-side orchestration
// ---------------------------------------------------------------------------
extern "C" void kernel_launch(void* const* d_in, const int* in_sizes, int n_in,
                              void* d_out, int out_size, void* d_ws, size_t ws_size,
                              hipStream_t stream) {
    (void)in_sizes; (void)n_in; (void)out_size; (void)ws_size;
    const float* query = (const float*)d_in[0];
    const float* key   = (const float*)d_in[1];
    const float* value = (const float*)d_in[2];
    const int*   mask  = (const int*)d_in[3];
    const float* Wq = (const float*)d_in[4];  const float* bq = (const float*)d_in[5];
    const float* Wk = (const float*)d_in[6];  const float* bk = (const float*)d_in[7];
    const float* Wv = (const float*)d_in[8];  const float* bv = (const float*)d_in[9];
    const float* Wo = (const float*)d_in[10]; const float* bo = (const float*)d_in[11];
    const float* ln_gamma = (const float*)d_in[12];
    const float* ln_beta  = (const float*)d_in[13];

    // Workspace carve-up (~80 MB total)
    char* p = (char*)d_ws;
    bf16_t* xq  = (bf16_t*)p; p += (size_t)M_ * D_ * sizeof(bf16_t);
    bf16_t* xk  = (bf16_t*)p; p += (size_t)M_ * D_ * sizeof(bf16_t);
    bf16_t* xv  = (bf16_t*)p; p += (size_t)M_ * D_ * sizeof(bf16_t);
    bf16_t* WqT = (bf16_t*)p; p += (size_t)D_ * D_ * sizeof(bf16_t);
    bf16_t* WkT = (bf16_t*)p; p += (size_t)D_ * D_ * sizeof(bf16_t);
    bf16_t* WvT = (bf16_t*)p; p += (size_t)D_ * D_ * sizeof(bf16_t);
    bf16_t* WoT = (bf16_t*)p; p += (size_t)D_ * D_ * sizeof(bf16_t);
    bf16_t* Qp  = (bf16_t*)p; p += (size_t)M_ * D_ * sizeof(bf16_t);   // [B][H][S][Dh]
    bf16_t* Kp  = (bf16_t*)p; p += (size_t)M_ * D_ * sizeof(bf16_t);   // [B][H][S][Dh]
    bf16_t* Vt  = (bf16_t*)p; p += (size_t)M_ * D_ * sizeof(bf16_t);   // [B][H][Dh][S]
    bf16_t* ctx = (bf16_t*)p; p += (size_t)M_ * D_ * sizeof(bf16_t);   // [B][S][D]
    float*  pre = (float*)p;  p += (size_t)M_ * D_ * sizeof(float);    // pre-LN

    float* out_ptr  = (float*)d_out;                    // [B][S][D]
    float* attn_ptr = out_ptr + (size_t)M_ * D_;        // [B][H][S][S]

    const int nElem = M_ * D_;
    cvt_f32_to_bf16_kernel<<<2048, 256, 0, stream>>>(query, xq, nElem);
    cvt_f32_to_bf16_kernel<<<2048, 256, 0, stream>>>(key,   xk, nElem);
    cvt_f32_to_bf16_kernel<<<2048, 256, 0, stream>>>(value, xv, nElem);

    dim3 tgrid(D_ / 32, D_ / 32), tblk(32, 8);
    transpose_cvt_kernel<<<tgrid, tblk, 0, stream>>>(Wq, WqT);
    transpose_cvt_kernel<<<tgrid, tblk, 0, stream>>>(Wk, WkT);
    transpose_cvt_kernel<<<tgrid, tblk, 0, stream>>>(Wv, WvT);
    transpose_cvt_kernel<<<tgrid, tblk, 0, stream>>>(Wo, WoT);

    dim3 pgrid(D_ / 128, M_ / 128);
    proj_gemm_kernel<<<pgrid, 256, 0, stream>>>(xq, WqT, bq, (void*)Qp, M_, D_, D_, 0);
    proj_gemm_kernel<<<pgrid, 256, 0, stream>>>(xk, WkT, bk, (void*)Kp, M_, D_, D_, 0);
    proj_gemm_kernel<<<pgrid, 256, 0, stream>>>(xv, WvT, bv, (void*)Vt, M_, D_, D_, 1);

    dim3 fgrid(S_ / 16, B_ * H_);
    fused_attn_kernel<<<fgrid, 256, 0, stream>>>(Qp, Kp, Vt, mask, attn_ptr, ctx);

    proj_gemm_kernel<<<pgrid, 256, 0, stream>>>(ctx, WoT, bo, (void*)pre, M_, D_, D_, 2);

    ln_kernel<<<M_, 256, 0, stream>>>(pre, query, ln_gamma, ln_beta, out_ptr);
}
